// AMFormerLayer_38190849196270
// MI455X (gfx1250) — matrix operand order repaired and checked by reading, not compiled, not run on hardware
//
#include <hip/hip_runtime.h>
#include <hip/hip_bf16.h>
#include <cstdint>
#include <cstddef>

// ---------------------------------------------------------------------------
// Problem constants (AMFormer layer)
// ---------------------------------------------------------------------------
#define BB   16
#define SSQ  512
#define DDM  768
#define HHD  12
#define DHH  64
#define DFF  3072
#define BSR  (BB * SSQ)          // 8192 rows of the flattened (B*S, D) activations

typedef __attribute__((ext_vector_type(16))) _Float16 v16h;
typedef __attribute__((ext_vector_type(8)))  float    v8f;
typedef int v4i __attribute__((vector_size(16)));     // matches builtin's V4i

#define AS1 __attribute__((address_space(1)))
#define AS3 __attribute__((address_space(3)))

// CDNA5 async global->LDS path (ASYNCcnt-tracked), with sync fallback.
#if __has_builtin(__builtin_amdgcn_global_load_async_to_lds_b128) && \
    __has_builtin(__builtin_amdgcn_s_wait_asynccnt)
#define USE_ASYNC_LDS 1
#else
#define USE_ASYNC_LDS 0
#endif

__device__ __forceinline__ void async_b128(const _Float16* g, _Float16* lds) {
#if USE_ASYNC_LDS
  __builtin_amdgcn_global_load_async_to_lds_b128(
      (AS1 v4i*)g, (AS3 v4i*)lds, 0, 0);
#else
  *reinterpret_cast<uint4*>(lds) = *reinterpret_cast<const uint4*>(g);
#endif
}

__device__ __forceinline__ void async_wait0() {
#if USE_ASYNC_LDS
  __builtin_amdgcn_s_wait_asynccnt(0);
#endif
}

// ---------------------------------------------------------------------------
// WMMA helpers (CDNA5: D = A(16x32 f16) x B(32x16 f16) + C(16x16 f32))
// ---------------------------------------------------------------------------
__device__ __forceinline__ v8f wmma16(v16h a, v16h b, v8f c) {
  return __builtin_amdgcn_wmma_f32_16x16x32_f16(
      /*neg_a=*/false, a, /*neg_b=*/false, b,
      /*c_mod=*/(short)0, c, /*reuse_a=*/false, /*reuse_b=*/false);
}

// A-matrix fragment: 16x32 f16, p points at a [16 x ld] tile (row = M, col = K).
__device__ __forceinline__ v16h load_a16(const _Float16* p, int ld) {
  const int lane = threadIdx.x & 31;
  const int hi = lane >> 4, m = lane & 15;
  union { v16h v; uint32_t u[8]; } f;
#pragma unroll
  for (int r = 0; r < 8; ++r) {
    const int k = (r < 4 ? (r << 1) : 16 + ((r - 4) << 1)) + (hi << 3);
    f.u[r] = *reinterpret_cast<const uint32_t*>(p + (size_t)m * ld + k);
  }
  return f.v;
}

// B-matrix fragment: 32x16 f16. p points at a [16 x ld] tile whose ROW index is
// the output column n and whose COLUMN index is the reduction index k.
__device__ __forceinline__ v16h load_b16(const _Float16* p, int ld) {
  const int lane = threadIdx.x & 31;
  const int hi = lane >> 4, n = lane & 15;
  union { v16h v; uint32_t u[8]; } f;
#pragma unroll
  for (int r = 0; r < 8; ++r) {
    const int k = (r < 4 ? (r << 1) : 16 + ((r - 4) << 1)) + (hi << 3);
    f.u[r] = *reinterpret_cast<const uint32_t*>(p + (size_t)n * ld + k);
  }
  return f.v;
}

// ---------------------------------------------------------------------------
// f32 -> f16 convert
// ---------------------------------------------------------------------------
__global__ void k_f32_to_f16(const float* __restrict__ src,
                             _Float16* __restrict__ dst, int n) {
  const int i = blockIdx.x * 256 + threadIdx.x;
  if (i < n) dst[i] = (_Float16)src[i];
}

// ---------------------------------------------------------------------------
// LayerNorm over last dim (768), fp32 in, f16 out
// ---------------------------------------------------------------------------
__global__ __launch_bounds__(256) void k_layernorm(
    const float* __restrict__ x, const float* __restrict__ g,
    const float* __restrict__ b, _Float16* __restrict__ out) {
  __shared__ float red[8];
  const int row = blockIdx.x;
  const float* xr = x + (size_t)row * DDM;

  float s = 0.f;
  for (int i = threadIdx.x; i < DDM; i += 256) s += xr[i];
  for (int off = 16; off; off >>= 1) s += __shfl_xor(s, off);
  if ((threadIdx.x & 31) == 0) red[threadIdx.x >> 5] = s;
  __syncthreads();
  float tot = 0.f;
#pragma unroll
  for (int w = 0; w < 8; ++w) tot += red[w];
  const float mu = tot * (1.f / DDM);
  __syncthreads();

  float v = 0.f;
  for (int i = threadIdx.x; i < DDM; i += 256) {
    const float d = xr[i] - mu;
    v += d * d;
  }
  for (int off = 16; off; off >>= 1) v += __shfl_xor(v, off);
  if ((threadIdx.x & 31) == 0) red[threadIdx.x >> 5] = v;
  __syncthreads();
  float tv = 0.f;
#pragma unroll
  for (int w = 0; w < 8; ++w) tv += red[w];
  const float rstd = rsqrtf(tv * (1.f / DDM) + 1e-5f);

  for (int i = threadIdx.x; i < DDM; i += 256)
    out[(size_t)row * DDM + i] = (_Float16)((xr[i] - mu) * rstd * g[i] + b[i]);
}

// ---------------------------------------------------------------------------
// Generic WMMA GEMM: C[M,N] = act(A[M,K] @ W[N,K]^T + bias)
// flags: bit0 = write f16, bit1 = exact GELU
// grid = (N/64, M/256), block = 256 (8 waves); wave w owns rows [w*32, w*32+32)
// Double-buffered LDS, async global->LDS staging overlapped with WMMA.
// ---------------------------------------------------------------------------
__global__ __launch_bounds__(256) void k_gemm_wmma(
    const _Float16* __restrict__ A, const _Float16* __restrict__ W,
    const float* __restrict__ bias, void* __restrict__ C,
    int M, int N, int K, int flags) {
  __shared__ __attribute__((aligned(16))) _Float16 sA[2][256][32];
  __shared__ __attribute__((aligned(16))) _Float16 sW[2][64][32];

  const int tid = threadIdx.x;
  const int lane = tid & 31, wave = tid >> 5;
  const int hi = lane >> 4, lm = lane & 15;
  const int bm = blockIdx.y * 256, bn = blockIdx.x * 64;

  // staging coordinates: A row per thread (halves 0-7 and 16-23), W 8 halves per thread
  const int wn = tid >> 2, wc = (tid & 3) << 3;

  auto stage = [&](int k0, int buf) {
    const _Float16* ga = A + (size_t)(bm + tid) * K + k0;
    async_b128(ga, &sA[buf][tid][0]);
    async_b128(ga + 16, &sA[buf][tid][16]);
    async_b128(W + (size_t)(bn + wn) * K + k0 + wc, &sW[buf][wn][wc]);
  };

  v8f acc[2][4] = {};   // [m-subtile][n-tile]
  const int T = K >> 5;
  stage(0, 0);
  for (int kt = 0; kt < T; ++kt) {
    async_wait0();        // this wave's tile-kt slices are in LDS
    __syncthreads();      // whole tile kt landed; other buffer free for reuse
    if (kt + 1 < T) stage((kt + 1) << 5, (kt + 1) & 1);
    const int buf = kt & 1;
    const v16h a0 = load_a16(&sA[buf][wave * 32][0], 32);
    const v16h a1 = load_a16(&sA[buf][wave * 32 + 16][0], 32);
#pragma unroll
    for (int t = 0; t < 4; ++t) {
      const v16h b = load_b16(&sW[buf][t * 16][0], 32);
      acc[0][t] = wmma16(a0, b, acc[0][t]);
      acc[1][t] = wmma16(a1, b, acc[1][t]);
    }
  }

#pragma unroll
  for (int mi = 0; mi < 2; ++mi)
#pragma unroll
    for (int t = 0; t < 4; ++t) {
      const int col = bn + t * 16 + lm;
      const float bv = bias[col];
#pragma unroll
      for (int r = 0; r < 8; ++r) {
        const int row = bm + wave * 32 + mi * 16 + r + (hi << 3);
        float v = acc[mi][t][r] + bv;
        if (flags & 2) v = 0.5f * v * (1.f + erff(v * 0.70710678118654752f));
        if (flags & 1)
          reinterpret_cast<_Float16*>(C)[(size_t)row * N + col] = (_Float16)v;
        else
          reinterpret_cast<float*>(C)[(size_t)row * N + col] = v;
      }
    }
}

// ---------------------------------------------------------------------------
// Flash attention over one head, streaming 64-key tiles.
// MODE 0: out = softmax(QK^T/8) @ V
// MODE 1: out = exp(P @ log|V|) * sign(P @ sign(V) + eps)   (P = softmax)
// grid = B*H*(S/64), block = 128 (4 waves; wave w -> 16 queries)
// ---------------------------------------------------------------------------
template <int MODE>
__global__ __launch_bounds__(128) void k_flash_attn(
    const _Float16* __restrict__ Q, const _Float16* __restrict__ Kx,
    const _Float16* __restrict__ Vx, _Float16* __restrict__ Out) {
  __shared__ __attribute__((aligned(16))) _Float16 sK[64][64];   // [key][d]
  __shared__ __attribute__((aligned(16))) _Float16 sV0[64][64];  // [d][key] (V or log|V|)
  __shared__ __attribute__((aligned(16))) _Float16 sV1[64][64];  // [d][key] (sign(V), MODE1)
  __shared__ __attribute__((aligned(16))) _Float16 sP[4][16][64];

  const int tid = threadIdx.x, lane = tid & 31, wave = tid >> 5;
  const int hi = lane >> 4, lm = lane & 15;
  const int qb = blockIdx.x & 7;            // S/64 = 8 query blocks
  const int bh = blockIdx.x >> 3;
  const int h = bh % HHD, b = bh / HHD;
  const size_t rowbase = (size_t)b * SSQ;
  const int q0 = qb * 64 + wave * 16;

  // Q fragments (two 16x32 K-chunks over DH=64), pre-scaled by 1/sqrt(DH)=0.125
  v16h aq0, aq1;
  {
    const _Float16* qp = Q + (rowbase + q0 + lm) * DDM + h * DHH;
    union { v16h v; _Float16 e[16]; } f0, f1;
#pragma unroll
    for (int r = 0; r < 8; ++r) {
      const int k = (r < 4 ? (r << 1) : 16 + ((r - 4) << 1)) + (hi << 3);
      f0.e[2 * r]     = (_Float16)((float)qp[k] * 0.125f);
      f0.e[2 * r + 1] = (_Float16)((float)qp[k + 1] * 0.125f);
      f1.e[2 * r]     = (_Float16)((float)qp[k + 32] * 0.125f);
      f1.e[2 * r + 1] = (_Float16)((float)qp[k + 33] * 0.125f);
    }
    aq0 = f0.v;
    aq1 = f1.v;
  }

  float m_run[8], l_run[8];
#pragma unroll
  for (int r = 0; r < 8; ++r) { m_run[r] = -1e30f; l_run[r] = 0.f; }
  v8f oa[4] = {};
  v8f ob[4] = {};

  for (int kt = 0; kt < SSQ / 64; ++kt) {
    __syncthreads();
    {
      const int r = tid >> 1, c = (tid & 1) << 5;
      // K tile: straight copy -> async path, overlaps with V transform below
      const _Float16* kg = Kx + (rowbase + kt * 64 + r) * DDM + h * DHH + c;
#pragma unroll
      for (int j = 0; j < 32; j += 8)
        async_b128(kg + j, &sK[r][c + j]);
      // V tile: transform + transpose on the VALU, ds_store path
      const _Float16* vg = Vx + (rowbase + kt * 64 + r) * DDM + h * DHH + c;
#pragma unroll
      for (int j = 0; j < 32; ++j) {
        const float fv = (float)vg[j];
        if (MODE == 0) {
          sV0[c + j][r] = (_Float16)fv;
        } else {
          sV0[c + j][r] = (_Float16)logf(fabsf(fv) + 1e-6f);
          sV1[c + j][r] = (_Float16)(fv > 0.f ? 1.f : (fv < 0.f ? -1.f : 0.f));
        }
      }
    }
    async_wait0();
    __syncthreads();

    // scores: 16 queries x 64 keys  (4 column tiles, 2 chained K-chunks each)
    v8f sc[4];
#pragma unroll
    for (int t = 0; t < 4; ++t) {
      v8f z = {};
      z = wmma16(aq0, load_b16(&sK[t * 16][0], 64), z);
      z = wmma16(aq1, load_b16(&sK[t * 16][32], 64), z);
      sc[t] = z;
    }

    // online softmax; row M = r + 8*hi lives across a 16-lane half-group
    float scl[8];
#pragma unroll
    for (int r = 0; r < 8; ++r) {
      float mx = fmaxf(fmaxf(sc[0][r], sc[1][r]), fmaxf(sc[2][r], sc[3][r]));
      mx = fmaxf(mx, __shfl_xor(mx, 1));
      mx = fmaxf(mx, __shfl_xor(mx, 2));
      mx = fmaxf(mx, __shfl_xor(mx, 4));
      mx = fmaxf(mx, __shfl_xor(mx, 8));
      const float mn = fmaxf(m_run[r], mx);
      scl[r] = expf(m_run[r] - mn);
      float rs = 0.f;
#pragma unroll
      for (int t = 0; t < 4; ++t) {
        const float p = expf(sc[t][r] - mn);
        sc[t][r] = p;
        rs += p;
      }
      rs += __shfl_xor(rs, 1);
      rs += __shfl_xor(rs, 2);
      rs += __shfl_xor(rs, 4);
      rs += __shfl_xor(rs, 8);
      l_run[r] = l_run[r] * scl[r] + rs;
      m_run[r] = mn;
    }
#pragma unroll
    for (int t = 0; t < 4; ++t)
#pragma unroll
      for (int r = 0; r < 8; ++r) {
        oa[t][r] *= scl[r];
        if (MODE == 1) ob[t][r] *= scl[r];
      }

    // P through per-wave LDS pad: C-layout -> A-fragment layout
#pragma unroll
    for (int t = 0; t < 4; ++t)
#pragma unroll
      for (int r = 0; r < 8; ++r)
        sP[wave][r + (hi << 3)][t * 16 + lm] = (_Float16)sc[t][r];

    const v16h ap0 = load_a16(&sP[wave][0][0], 64);
    const v16h ap1 = load_a16(&sP[wave][0][32], 64);
#pragma unroll
    for (int t = 0; t < 4; ++t) {
      oa[t] = wmma16(ap0, load_b16(&sV0[t * 16][0], 64), oa[t]);
      oa[t] = wmma16(ap1, load_b16(&sV0[t * 16][32], 64), oa[t]);
      if (MODE == 1) {
        ob[t] = wmma16(ap0, load_b16(&sV1[t * 16][0], 64), ob[t]);
        ob[t] = wmma16(ap1, load_b16(&sV1[t * 16][32], 64), ob[t]);
      }
    }
  }

  _Float16* op = Out + (rowbase + q0) * DDM + h * DHH;
#pragma unroll
  for (int t = 0; t < 4; ++t)
#pragma unroll
    for (int r = 0; r < 8; ++r) {
      const int row = r + (hi << 3);
      const float inv = 1.f / l_run[r];
      float val;
      if (MODE == 0) {
        val = oa[t][r] * inv;
      } else {
        const float lo = oa[t][r] * inv;
        const float sw = ob[t][r] * inv + 1e-6f;
        const float sg = sw > 0.f ? 1.f : (sw < 0.f ? -1.f : 0.f);
        val = expf(lo) * sg;
      }
      op[(size_t)row * DDM + t * 16 + lm] = (_Float16)val;
    }
}

// ---------------------------------------------------------------------------
// elementwise: x2 = x + alpha*add_out + (1-alpha)*mul_out
// ---------------------------------------------------------------------------
__global__ void k_combine(const float* __restrict__ x,
                          const float* __restrict__ addo,
                          const float* __restrict__ mulo,
                          const float* __restrict__ mixw,
                          float* __restrict__ out, int n) {
  const int i = blockIdx.x * 256 + threadIdx.x;
  if (i < n) {
    const float a = 1.f / (1.f + expf(-mixw[0]));
    out[i] = x[i] + a * addo[i] + (1.f - a) * mulo[i];
  }
}

__global__ void k_add_inplace(float* __restrict__ out,
                              const float* __restrict__ t, int n) {
  const int i = blockIdx.x * 256 + threadIdx.x;
  if (i < n) out[i] += t[i];
}

// ---------------------------------------------------------------------------
// Launch
// ---------------------------------------------------------------------------
extern "C" void kernel_launch(void* const* d_in, const int* in_sizes, int n_in,
                              void* d_out, int out_size, void* d_ws, size_t ws_size,
                              hipStream_t stream) {
  (void)in_sizes; (void)n_in; (void)out_size; (void)ws_size;

  const float* x       = (const float*)d_in[0];
  const float* Wqa     = (const float*)d_in[1];
  const float* bqa     = (const float*)d_in[2];
  const float* Wka     = (const float*)d_in[3];
  const float* bka     = (const float*)d_in[4];
  const float* Wva     = (const float*)d_in[5];
  const float* bva     = (const float*)d_in[6];
  const float* Woa     = (const float*)d_in[7];
  const float* boa     = (const float*)d_in[8];
  const float* Wqm     = (const float*)d_in[9];
  const float* bqm     = (const float*)d_in[10];
  const float* Wkm     = (const float*)d_in[11];
  const float* bkm     = (const float*)d_in[12];
  const float* Wvm     = (const float*)d_in[13];
  const float* bvm     = (const float*)d_in[14];
  const float* Wom     = (const float*)d_in[15];
  const float* bom     = (const float*)d_in[16];
  const float* Wf1     = (const float*)d_in[17];
  const float* bf1     = (const float*)d_in[18];
  const float* Wf2     = (const float*)d_in[19];
  const float* bf2     = (const float*)d_in[20];
  const float* ln1g    = (const float*)d_in[21];
  const float* ln1b    = (const float*)d_in[22];
  const float* ln2g    = (const float*)d_in[23];
  const float* ln2b    = (const float*)d_in[24];
  const float* mixw    = (const float*)d_in[25];

  // ----- workspace layout (256B aligned slices) -----
  char* ws = (char*)d_ws;
  size_t off = 0;
  auto take = [&](size_t bytes) {
    size_t o = off;
    off += (bytes + 255) & ~(size_t)255;
    return o;
  };
  constexpr size_t EH = sizeof(_Float16);
  const size_t nAct = (size_t)BSR * DDM;       // 8192*768
  const size_t nPW  = (size_t)DDM * DDM;       // 768*768
  const size_t nFW  = (size_t)DFF * DDM;       // 3072*768

  const size_t oXn   = take(nAct * EH);        // LN1 out (reused as LN2 out)
  const size_t oQa   = take(nAct * EH);        // also start of FFN1-out reuse region
  const size_t oKa   = take(nAct * EH);
  const size_t oVa   = take(nAct * EH);
  const size_t oQm   = take(nAct * EH);
  const size_t oKm   = take(nAct * EH);
  const size_t oVm   = take(nAct * EH);
  const size_t oCa   = take(nAct * EH);        // attention core (add)
  const size_t oCm   = take(nAct * EH);        // attention core (mul)
  const size_t oAo   = take(nAct * 4);         // add_out f32 (reused as ffn2 tmp)
  const size_t oMo   = take(nAct * 4);         // mul_out f32
  const size_t oWqa  = take(nPW * EH);
  const size_t oWka  = take(nPW * EH);
  const size_t oWva  = take(nPW * EH);
  const size_t oWoa  = take(nPW * EH);
  const size_t oWqm  = take(nPW * EH);
  const size_t oWkm  = take(nPW * EH);
  const size_t oWvm  = take(nPW * EH);
  const size_t oWom  = take(nPW * EH);
  const size_t oWf1  = take(nFW * EH);
  const size_t oWf2  = take(nFW * EH);

  _Float16* hXn = (_Float16*)(ws + oXn);
  _Float16* hQa = (_Float16*)(ws + oQa);
  _Float16* hKa = (_Float16*)(ws + oKa);
  _Float16* hVa = (_Float16*)(ws + oVa);
  _Float16* hQm = (_Float16*)(ws + oQm);
  _Float16* hKm = (_Float16*)(ws + oKm);
  _Float16* hVm = (_Float16*)(ws + oVm);
  _Float16* hCa = (_Float16*)(ws + oCa);
  _Float16* hCm = (_Float16*)(ws + oCm);
  float*    fAo = (float*)(ws + oAo);
  float*    fMo = (float*)(ws + oMo);
  _Float16* hF1 = hQa;                 // reuse q/k/v region for FFN1 output (50MB < 75MB)
  float*    fT2 = fAo;                 // reuse add_out for FFN2 output
  _Float16* hH2 = hXn;                 // reuse LN1 buffer for LN2 output

  _Float16* hWqa = (_Float16*)(ws + oWqa);
  _Float16* hWka = (_Float16*)(ws + oWka);
  _Float16* hWva = (_Float16*)(ws + oWva);
  _Float16* hWoa = (_Float16*)(ws + oWoa);
  _Float16* hWqm = (_Float16*)(ws + oWqm);
  _Float16* hWkm = (_Float16*)(ws + oWkm);
  _Float16* hWvm = (_Float16*)(ws + oWvm);
  _Float16* hWom = (_Float16*)(ws + oWom);
  _Float16* hWf1 = (_Float16*)(ws + oWf1);
  _Float16* hWf2 = (_Float16*)(ws + oWf2);

  // ----- 1. convert weights to f16 -----
  const int gPW = (int)((nPW + 255) / 256), gFW = (int)((nFW + 255) / 256);
  k_f32_to_f16<<<gPW, 256, 0, stream>>>(Wqa, hWqa, (int)nPW);
  k_f32_to_f16<<<gPW, 256, 0, stream>>>(Wka, hWka, (int)nPW);
  k_f32_to_f16<<<gPW, 256, 0, stream>>>(Wva, hWva, (int)nPW);
  k_f32_to_f16<<<gPW, 256, 0, stream>>>(Woa, hWoa, (int)nPW);
  k_f32_to_f16<<<gPW, 256, 0, stream>>>(Wqm, hWqm, (int)nPW);
  k_f32_to_f16<<<gPW, 256, 0, stream>>>(Wkm, hWkm, (int)nPW);
  k_f32_to_f16<<<gPW, 256, 0, stream>>>(Wvm, hWvm, (int)nPW);
  k_f32_to_f16<<<gPW, 256, 0, stream>>>(Wom, hWom, (int)nPW);
  k_f32_to_f16<<<gFW, 256, 0, stream>>>(Wf1, hWf1, (int)nFW);
  k_f32_to_f16<<<gFW, 256, 0, stream>>>(Wf2, hWf2, (int)nFW);

  // ----- 2. LN1 -----
  k_layernorm<<<BSR, 256, 0, stream>>>(x, ln1g, ln1b, hXn);

  // ----- 3. QKV projections (both branches), f16 outputs -----
  const dim3 gProj(DDM / 64, BSR / 256);
  k_gemm_wmma<<<gProj, 256, 0, stream>>>(hXn, hWqa, bqa, hQa, BSR, DDM, DDM, 1);
  k_gemm_wmma<<<gProj, 256, 0, stream>>>(hXn, hWka, bka, hKa, BSR, DDM, DDM, 1);
  k_gemm_wmma<<<gProj, 256, 0, stream>>>(hXn, hWva, bva, hVa, BSR, DDM, DDM, 1);
  k_gemm_wmma<<<gProj, 256, 0, stream>>>(hXn, hWqm, bqm, hQm, BSR, DDM, DDM, 1);
  k_gemm_wmma<<<gProj, 256, 0, stream>>>(hXn, hWkm, bkm, hKm, BSR, DDM, DDM, 1);
  k_gemm_wmma<<<gProj, 256, 0, stream>>>(hXn, hWvm, bvm, hVm, BSR, DDM, DDM, 1);

  // ----- 4. attention (flash style, WMMA) -----
  const int gAttn = BB * HHD * (SSQ / 64);   // 1536 blocks
  k_flash_attn<0><<<gAttn, 128, 0, stream>>>(hQa, hKa, hVa, hCa);
  k_flash_attn<1><<<gAttn, 128, 0, stream>>>(hQm, hKm, hVm, hCm);

  // ----- 5. output projections, f32 outputs -----
  k_gemm_wmma<<<gProj, 256, 0, stream>>>(hCa, hWoa, boa, fAo, BSR, DDM, DDM, 0);
  k_gemm_wmma<<<gProj, 256, 0, stream>>>(hCm, hWom, bom, fMo, BSR, DDM, DDM, 0);

  // ----- 6. mix + residual -> d_out holds x2 -----
  const int nE = BSR * DDM;
  k_combine<<<nE / 256, 256, 0, stream>>>(x, fAo, fMo, mixw, (float*)d_out, nE);

  // ----- 7. LN2 + FFN -----
  k_layernorm<<<BSR, 256, 0, stream>>>((const float*)d_out, ln2g, ln2b, hH2);
  const dim3 gF1(DFF / 64, BSR / 256);
  k_gemm_wmma<<<gF1, 256, 0, stream>>>(hH2, hWf1, bf1, hF1, BSR, DFF, DDM, 3);  // f16 + GELU
  const dim3 gF2(DDM / 64, BSR / 256);
  k_gemm_wmma<<<gF2, 256, 0, stream>>>(hF1, hWf2, bf2, fT2, BSR, DDM, DFF, 0);

  // ----- 8. final residual -----
  k_add_inplace<<<nE / 256, 256, 0, stream>>>((float*)d_out, fT2, nE);
}
